// PCModel_17952963297900
// MI455X (gfx1250) — matrix operand order repaired and checked by reading, not compiled, not run on hardware
//
#include <hip/hip_runtime.h>
#include <hip/hip_bf16.h>

// ---------------------------------------------------------------------------
// PointNet++ forward for MI455X (gfx1250, wave32).
// - All 1x1-conv matmuls run on v_wmma_f32_16x16x32_f16 with branch-free
//   global_load_b128 fragment loads from pre-padded f16 operand buffers.
// - Ball-query: wave32 ballot ordered scan, writes padded-f16 grouped tensor.
// - FPS: serial bottleneck; wave shfl_xor butterfly + one LDS pass per step.
// ---------------------------------------------------------------------------

typedef __attribute__((ext_vector_type(16))) _Float16 v16h;
typedef __attribute__((ext_vector_type(8)))  _Float16 v8h;
typedef __attribute__((ext_vector_type(8)))  float    v8f;

union V16U { v16h v; struct { v8h lo; v8h hi; } p; };

// ------------------------- transpose [C,N] -> [N,C] -------------------------
__global__ void k_transpose_cn(const float* __restrict__ src, float* __restrict__ dst,
                               int C, int N) {
  int idx = blockIdx.x * blockDim.x + threadIdx.x;
  if (idx >= C * N) return;
  int c = idx / N, n = idx % N;
  dst[(size_t)n * C + c] = src[idx];
}

// ------------------------- final [N,C] -> [C,N] -----------------------------
__global__ void k_out_transpose(const float* __restrict__ src, float* __restrict__ dst,
                                int N, int C) {
  int idx = blockIdx.x * blockDim.x + threadIdx.x;
  if (idx >= C * N) return;
  int n = idx / C, c = idx % C;
  dst[(size_t)c * N + n] = src[idx];
}

// ------------------------- farthest point sampling --------------------------
#define FPS_T 512
__global__ __launch_bounds__(FPS_T)
void k_fps(const float* __restrict__ xyz, int N, int S,
           int* __restrict__ fidx, float* __restrict__ newxyz,
           float* __restrict__ dist) {
  __shared__ float s_val[FPS_T / 32];
  __shared__ int   s_idx[FPS_T / 32];
  __shared__ float s_c[3];
  __shared__ int   s_far;
  int t = threadIdx.x;
  for (int i = t; i < N; i += FPS_T) dist[i] = 1e10f;
  if (t == 0) s_far = 0;
  __syncthreads();
  for (int it = 0; it < S; ++it) {
    if (t == 0) {
      int far = s_far;
      fidx[it] = far;
      s_c[0] = xyz[far * 3 + 0];
      s_c[1] = xyz[far * 3 + 1];
      s_c[2] = xyz[far * 3 + 2];
      newxyz[it * 3 + 0] = s_c[0];
      newxyz[it * 3 + 1] = s_c[1];
      newxyz[it * 3 + 2] = s_c[2];
    }
    __syncthreads();
    float cx = s_c[0], cy = s_c[1], cz = s_c[2];
    float best = -1.0f; int bi = N - 1;
    for (int i = t; i < N; i += FPS_T) {
      float dx = xyz[i * 3 + 0] - cx;
      float dy = xyz[i * 3 + 1] - cy;
      float dz = xyz[i * 3 + 2] - cz;
      float d = dx * dx + dy * dy + dz * dz;
      float dd = fminf(dist[i], d);
      dist[i] = dd;
      if (dd > best) { best = dd; bi = i; }
    }
    // wave32 butterfly argmax (first-index tie-break)
#pragma unroll
    for (int off = 16; off > 0; off >>= 1) {
      float ov = __shfl_xor(best, off, 32);
      int   oi = __shfl_xor(bi,   off, 32);
      if (ov > best || (ov == best && oi < bi)) { best = ov; bi = oi; }
    }
    int w = t >> 5;
    if ((t & 31) == 0) { s_val[w] = best; s_idx[w] = bi; }
    __syncthreads();
    if (t == 0) {
      float bb = s_val[0]; int ii = s_idx[0];
      for (int q = 1; q < FPS_T / 32; ++q) {
        if (s_val[q] > bb || (s_val[q] == bb && s_idx[q] < ii)) {
          bb = s_val[q]; ii = s_idx[q];
        }
      }
      s_far = ii;
    }
    __syncthreads();
  }
}

// ---- ball query (first-K in-radius ascending) + group -> padded f16 rows ---
__global__ __launch_bounds__(32)
void k_query_group_f16(const float* __restrict__ xyz, const float* __restrict__ pts,
                       const float* __restrict__ newxyz, int N, int K, int Cp,
                       float r2, int Kp, _Float16* __restrict__ g) {
  int s = blockIdx.x;
  int lane = threadIdx.x;
  float cx = newxyz[s * 3 + 0];
  float cy = newxyz[s * 3 + 1];
  float cz = newxyz[s * 3 + 2];
  __shared__ int idxbuf[64];
  int cnt = 0;
  for (int base = 0; base < N && cnt < K; base += 32) {
    int j = base + lane;
    bool in = false;
    if (j < N) {
      float dx = xyz[j * 3 + 0] - cx;
      float dy = xyz[j * 3 + 1] - cy;
      float dz = xyz[j * 3 + 2] - cz;
      in = (dx * dx + dy * dy + dz * dz) <= r2;
    }
    unsigned mask = (unsigned)__ballot(in);
    if (in) {
      int rank = __popc(mask & ((1u << lane) - 1u));
      int pos = cnt + rank;
      if (pos < K) idxbuf[pos] = j;
    }
    cnt += __popc(mask);
  }
  __syncthreads();
  if (cnt > K) cnt = K;
  int Ct = 3 + Cp;
  for (int k = lane; k < K; k += 32) {
    int j = (k < cnt) ? idxbuf[k] : (N - 1);
    _Float16* o = g + ((size_t)s * K + k) * Kp;
    o[0] = (_Float16)(xyz[j * 3 + 0] - cx);
    o[1] = (_Float16)(xyz[j * 3 + 1] - cy);
    o[2] = (_Float16)(xyz[j * 3 + 2] - cz);
    for (int c = 0; c < Cp; ++c) o[3 + c] = (_Float16)pts[(size_t)j * Cp + c];
    for (int c = Ct; c < Kp; ++c) o[c] = (_Float16)0.f;
  }
}

// ---------- f32 [M,K] -> zero-padded f16 [Mp,Kp] (rows/cols padded) ---------
__global__ void k_cvt_pad_f16(const float* __restrict__ src, _Float16* __restrict__ dst,
                              int M, int K, int Mp, int Kp) {
  int idx = blockIdx.x * blockDim.x + threadIdx.x;
  if (idx >= Mp * Kp) return;
  int r = idx / Kp, c = idx % Kp;
  float v = (r < M && c < K) ? src[(size_t)r * K + c] : 0.f;
  dst[idx] = (_Float16)v;
}

// ---------------- WMMA GEMM: Y[M,N] = Xh[M,Kp] * Wh[Np,Kp]^T + bias ---------
// Padded f16 operands -> branch-free b128 fragment loads, f32 accumulation.
__global__ __launch_bounds__(32)
void k_gemm_wmma(const _Float16* __restrict__ Xh, const _Float16* __restrict__ Wh,
                 const float* __restrict__ bias, float* __restrict__ Y,
                 int M, int N, int Kp) {
  const int lane = threadIdx.x & 31;
  const int half = lane >> 4;
  const int l15  = lane & 15;
  const int m0 = blockIdx.x * 16;
  const int n0 = blockIdx.y * 16;

  const _Float16* xr = Xh + (size_t)(m0 + l15) * Kp;  // A row (lanes 0-15 == 16-31)
  const _Float16* wr = Wh + (size_t)(n0 + l15) * Kp;  // B column = W row

  v8f acc = {};
  for (int kb = 0; kb < Kp; kb += 32) {
    V16U A, Bm;
    // A 16x32 f16: lane half selects K sub-block; two contiguous 8-half chunks
    A.p.lo = *(const v8h*)(xr + kb + half * 8);
    A.p.hi = *(const v8h*)(xr + kb + 16 + half * 8);
    // B 32x16 f16: same K order as A (lane half selects 8-wide K sub-blocks)
    Bm.p.lo = *(const v8h*)(wr + kb + half * 8);
    Bm.p.hi = *(const v8h*)(wr + kb + 16 + half * 8);
    acc = __builtin_amdgcn_wmma_f32_16x16x32_f16(
        /*neg_a=*/false, A.v, /*neg_b=*/false, Bm.v,
        /*c_mod=*/(short)0, acc, /*reuse_a=*/false, /*reuse_b=*/false);
  }
  const int nOut = n0 + l15;
  float bv = (nOut < N) ? bias[nOut] : 0.f;
#pragma unroll
  for (int j = 0; j < 8; ++j) {
    int m = m0 + half * 8 + j;
    if (m < M && nOut < N) Y[(size_t)m * N + nOut] = acc[j] + bv;
  }
}

// ------------------------- batchnorm statistics -----------------------------
__global__ void k_bn_stats(const float* __restrict__ x, int M, int C,
                           float* __restrict__ mean, float* __restrict__ var) {
  __shared__ float ssum[256], ssq[256];
  int c = blockIdx.x;
  float sum = 0.f, sq = 0.f;
  for (int i = threadIdx.x; i < M; i += 256) {
    float v = x[(size_t)i * C + c];
    sum += v; sq += v * v;
  }
  ssum[threadIdx.x] = sum; ssq[threadIdx.x] = sq;
  __syncthreads();
  for (int off = 128; off > 0; off >>= 1) {
    if (threadIdx.x < off) {
      ssum[threadIdx.x] += ssum[threadIdx.x + off];
      ssq[threadIdx.x]  += ssq[threadIdx.x + off];
    }
    __syncthreads();
  }
  if (threadIdx.x == 0) {
    float m = ssum[0] / (float)M;
    mean[c] = m;
    var[c]  = ssq[0] / (float)M - m * m;
  }
}

// ------- BN apply + ReLU (in-place f32) fused with padded-f16 convert -------
__global__ void k_bn_relu_cvt(float* __restrict__ x, int M, int C, int Kp,
                              const float* __restrict__ mean, const float* __restrict__ var,
                              const float* __restrict__ gam, const float* __restrict__ bet,
                              _Float16* __restrict__ xh) {
  int idx = blockIdx.x * blockDim.x + threadIdx.x;
  if (idx >= M * Kp) return;
  int r = idx / Kp, c = idx % Kp;
  if (c < C) {
    size_t fi = (size_t)r * C + c;
    float v = (x[fi] - mean[c]) * rsqrtf(var[c] + 1e-5f) * gam[c] + bet[c];
    v = v > 0.f ? v : 0.f;
    x[fi] = v;
    xh[idx] = (_Float16)v;
  } else {
    xh[idx] = (_Float16)0.f;
  }
}

// ------------------------- max pool over K ----------------------------------
__global__ void k_maxpool(const float* __restrict__ x, int S, int K, int C,
                          float* __restrict__ y) {
  int idx = blockIdx.x * blockDim.x + threadIdx.x;
  if (idx >= S * C) return;
  int s = idx / C, c = idx % C;
  float m = -3.4e38f;
  for (int k = 0; k < K; ++k) m = fmaxf(m, x[((size_t)s * K + k) * C + c]);
  y[idx] = m;
}

// ------------------------- copy columns (concat) ----------------------------
__global__ void k_copy_cols(const float* __restrict__ src, float* __restrict__ dst,
                            int N, int Cs, int Ct, int coff) {
  int idx = blockIdx.x * blockDim.x + threadIdx.x;
  if (idx >= N * Cs) return;
  int n = idx / Cs, c = idx % Cs;
  dst[(size_t)n * Ct + coff + c] = src[idx];
}

// ------------------------- 3-NN interpolation -------------------------------
__global__ void k_three_interp(const float* __restrict__ xyz1, const float* __restrict__ xyz2,
                               const float* __restrict__ pts2, int N1, int N2, int C2,
                               float* __restrict__ out, int Ct, int coff) {
  int n = blockIdx.x * blockDim.x + threadIdx.x;
  if (n >= N1) return;
  float x = xyz1[n * 3 + 0], y = xyz1[n * 3 + 1], z = xyz1[n * 3 + 2];
  float d0 = 3.4e38f, d1 = 3.4e38f, d2 = 3.4e38f;
  int i0 = 0, i1 = 0, i2 = 0;
  for (int j = 0; j < N2; ++j) {
    float dx = x - xyz2[j * 3 + 0];
    float dy = y - xyz2[j * 3 + 1];
    float dz = z - xyz2[j * 3 + 2];
    float d = dx * dx + dy * dy + dz * dz;
    if (d < d0)      { d2 = d1; i2 = i1; d1 = d0; i1 = i0; d0 = d; i0 = j; }
    else if (d < d1) { d2 = d1; i2 = i1; d1 = d; i1 = j; }
    else if (d < d2) { d2 = d; i2 = j; }
  }
  float w0 = 1.f / (d0 + 1e-8f);
  float w1 = 1.f / (d1 + 1e-8f);
  float w2 = 1.f / (d2 + 1e-8f);
  float ws = w0 + w1 + w2;
  w0 /= ws; w1 /= ws; w2 /= ws;
  float* o = out + (size_t)n * Ct + coff;
  const float* p0 = pts2 + (size_t)i0 * C2;
  const float* p1 = pts2 + (size_t)i1 * C2;
  const float* p2 = pts2 + (size_t)i2 * C2;
  for (int c = 0; c < C2; ++c) o[c] = w0 * p0[c] + w1 * p1[c] + w2 * p2[c];
}

// ---------------------------------------------------------------------------
static inline int pad16(int x) { return (x + 15) & ~15; }
static inline int pad32(int x) { return (x + 31) & ~31; }

extern "C" void kernel_launch(void* const* d_in, const int* in_sizes, int n_in,
                              void* d_out, int out_size, void* d_ws, size_t ws_size,
                              hipStream_t stream) {
  (void)in_sizes; (void)n_in; (void)out_size; (void)ws_size;

  static const int   LN[6]       = {16384, 4096, 1024, 256, 64, 16};
  static const int   LC[6]       = {4, 8, 16, 32, 64, 128};
  static const int   SA_K[5]     = {32, 16, 8, 4, 2};
  static const float SA_R[5]     = {0.05f, 0.1f, 0.2f, 0.4f, 0.8f};
  static const int   SA_INC[5]   = {7, 11, 19, 35, 67};
  static const int   SA_MLP[5][3]= {{4,4,8},{8,8,16},{16,16,32},{32,32,64},{64,64,128}};
  static const int   FP_OUT[5]   = {64, 32, 16, 8, 8};

  // ---- parse inputs (dict order, nested lists flattened) ----
  const float* xyz_in = (const float*)d_in[0];   // [3,16384]
  const float* pts_in = (const float*)d_in[1];   // [4,16384]
  int p = 2;
  const float *saW[5][3], *saB[5][3], *saG[5][3], *saBt[5][3];
  for (int L = 0; L < 5; ++L)
    for (int j = 0; j < 3; ++j) {
      saW[L][j]  = (const float*)d_in[p++];
      saB[L][j]  = (const float*)d_in[p++];
      saG[L][j]  = (const float*)d_in[p++];
      saBt[L][j] = (const float*)d_in[p++];
    }
  const float *fpW[5], *fpB[5], *fpG[5], *fpBt[5];
  for (int t = 0; t < 5; ++t) {
    fpW[t]  = (const float*)d_in[p++];
    fpB[t]  = (const float*)d_in[p++];
    fpG[t]  = (const float*)d_in[p++];
    fpBt[t] = (const float*)d_in[p++];
  }

  // ---- bump-allocate workspace ----
  char* ws = (char*)d_ws;
  size_t off = 0;
  auto allocf = [&](size_t elems) -> float* {
    float* ptr = (float*)(ws + off);
    off = (off + elems * sizeof(float) + 255) & ~(size_t)255;
    return ptr;
  };
  auto alloch = [&](size_t elems) -> _Float16* {
    _Float16* ptr = (_Float16*)(ws + off);
    off = (off + elems * sizeof(_Float16) + 255) & ~(size_t)255;
    return ptr;
  };
  float* lx[6]; float* lp[6];
  for (int i = 0; i < 6; ++i) lx[i] = allocf((size_t)LN[i] * 3);
  for (int i = 0; i < 6; ++i) lp[i] = allocf((size_t)LN[i] * LC[i]);
  float*    dist  = allocf(16384);
  int*      fidx  = (int*)allocf(4096);
  float*    sa_a  = allocf((size_t)4096 * 32 * 8);   // f32 conv outputs (ping)
  float*    sa_b  = allocf((size_t)4096 * 32 * 8);   // (pong)
  float*    meanb = allocf(256);
  float*    varb  = allocf(256);
  float*    fpxin = allocf((size_t)16384 * 8 + 64 * 192);
  float*    fp_fa = allocf((size_t)16384 * 8);
  float*    fp_fb = allocf((size_t)16384 * 8);
  _Float16* xh    = alloch((size_t)131072 * 32);     // padded f16 activations
  _Float16* wh    = alloch(32768);                   // padded f16 weights

  // ---- input transposes: [C,N] -> [N,C] ----
  {
    int tot = 3 * 16384;
    k_transpose_cn<<<(tot + 255) / 256, 256, 0, stream>>>(xyz_in, lx[0], 3, 16384);
    tot = 4 * 16384;
    k_transpose_cn<<<(tot + 255) / 256, 256, 0, stream>>>(pts_in, lp[0], 4, 16384);
  }

  // ---- SA levels ----
  for (int L = 0; L < 5; ++L) {
    int Nl = LN[L], S = LN[L + 1], K = SA_K[L], Cp = LC[L];
    k_fps<<<1, FPS_T, 0, stream>>>(lx[L], Nl, S, fidx, lx[L + 1], dist);
    float r2 = SA_R[L] * SA_R[L];
    int ci = SA_INC[L];
    int Kp = pad32(ci);
    k_query_group_f16<<<S, 32, 0, stream>>>(lx[L], lp[L], lx[L + 1], Nl, K, Cp,
                                            r2, Kp, xh);
    int M = S * K;   // multiple of 16 at every level
    float* dstf = sa_a;
    for (int j = 0; j < 3; ++j) {
      int co = SA_MLP[L][j];
      int Np = pad16(co);
      // weights -> padded f16
      {
        int tot = Np * Kp;
        k_cvt_pad_f16<<<(tot + 255) / 256, 256, 0, stream>>>(saW[L][j], wh, co, ci, Np, Kp);
      }
      dim3 g(M / 16, Np / 16);
      k_gemm_wmma<<<g, 32, 0, stream>>>(xh, wh, saB[L][j], dstf, M, co, Kp);
      k_bn_stats<<<co, 256, 0, stream>>>(dstf, M, co, meanb, varb);
      int KpN = pad32(co);
      int tot = M * KpN;
      k_bn_relu_cvt<<<(tot + 255) / 256, 256, 0, stream>>>(dstf, M, co, KpN,
                                                           meanb, varb, saG[L][j],
                                                           saBt[L][j], xh);
      ci = co; Kp = KpN;
      dstf = (dstf == sa_a) ? sa_b : sa_a;
    }
    // final activations (f32) are in the buffer written by conv3
    const float* act = (dstf == sa_a) ? sa_b : sa_a;
    int tot = S * ci;
    k_maxpool<<<(tot + 255) / 256, 256, 0, stream>>>(act, S, K, ci, lp[L + 1]);
  }

  // ---- FP levels ----
  const float* feats = lp[5];
  int Cf = 128;
  float* fa = fp_fa;
  float* fb = fp_fb;
  for (int t = 0; t < 5; ++t) {
    int lvl = 4 - t;
    int N1 = LN[lvl], N2 = LN[lvl + 1];
    int C1 = (lvl > 0) ? LC[lvl] : 0;
    int Ct = C1 + Cf;
    if (C1 > 0) {
      int tot = N1 * C1;
      k_copy_cols<<<(tot + 255) / 256, 256, 0, stream>>>(lp[lvl], fpxin, N1, C1, Ct, 0);
    }
    k_three_interp<<<(N1 + 255) / 256, 256, 0, stream>>>(lx[lvl], lx[lvl + 1], feats,
                                                         N1, N2, Cf, fpxin, Ct, C1);
    int Kp = pad32(Ct);
    {
      int tot = N1 * Kp;
      k_cvt_pad_f16<<<(tot + 255) / 256, 256, 0, stream>>>(fpxin, xh, N1, Ct, N1, Kp);
    }
    int co = FP_OUT[t];
    int Np = pad16(co);
    {
      int tot = Np * Kp;
      k_cvt_pad_f16<<<(tot + 255) / 256, 256, 0, stream>>>(fpW[t], wh, co, Ct, Np, Kp);
    }
    dim3 g(N1 / 16, Np / 16);
    k_gemm_wmma<<<g, 32, 0, stream>>>(xh, wh, fpB[t], fa, N1, co, Kp);
    k_bn_stats<<<co, 256, 0, stream>>>(fa, N1, co, meanb, varb);
    {
      // apply BN+ReLU in place (f16 copy unused after last level but harmless)
      int KpN = pad32(co);
      int tot = N1 * KpN;
      k_bn_relu_cvt<<<(tot + 255) / 256, 256, 0, stream>>>(fa, N1, co, KpN,
                                                           meanb, varb, fpG[t],
                                                           fpBt[t], xh);
    }
    feats = fa;
    Cf = co;
    float* tmp = fa; fa = fb; fb = tmp;
  }

  // ---- output transpose [16384,8] -> [8,16384] ----
  {
    int tot = 16384 * 8;
    k_out_transpose<<<(tot + 255) / 256, 256, 0, stream>>>(feats, (float*)d_out, 16384, 8);
  }
}
